// RecursiveReasoningModule_36687610642943
// MI455X (gfx1250) — compile-verified, matrix operand order checked by reading, not run on hardware
//
#include <hip/hip_runtime.h>

// ---------------------------------------------------------------------------
// RecursiveReasoningModule for MI455X (gfx1250, wave32, WMMA).
// Pipeline: mean-pool (HBM-bound, 256MB) -> bf16 weight transposes ->
// chain of small bf16 WMMA GEMMs (fp32 accum) with fused bias/GELU/residual.
// Attention with S=1 collapses to o=v, so q/k projections are skipped.
// ---------------------------------------------------------------------------

typedef __attribute__((ext_vector_type(16))) __bf16 v16bf;
typedef __attribute__((ext_vector_type(8)))  float  v8f;

#define M_ROWS 128
#define HID    1024
#define RD     512
#define AD     256
#define FF     2048
#define NLAYER 2
#define NREC   6

__device__ __forceinline__ unsigned short f2bf(float f) {
    unsigned int u = __float_as_uint(f);
    u += 0x7FFFu + ((u >> 16) & 1u);   // round to nearest even
    return (unsigned short)(u >> 16);
}

union Frag { unsigned int u[8]; v16bf v; };

// ---------------------------------------------------------------------------
// GEMM: C[M,N] = epilogue(A[M,K](bf16) @ Bt[N,K](bf16)^T + bias [+ Cin])
// One wave per 16x16 tile, 4 waves (64 cols) per block. K % 32 == 0.
// ACT: 0 = none, 1 = tanh-GELU (applied before residual add, matching ref).
// ---------------------------------------------------------------------------
template<int ACT>
__global__ __launch_bounds__(128)
void gemm_bf16_wmma(const unsigned short* __restrict__ A, int lda,
                    const unsigned short* __restrict__ Bt, int ldb,
                    const float* __restrict__ bias,
                    const float* __restrict__ Cin,
                    float* __restrict__ Cf,
                    unsigned short* __restrict__ Cb,
                    int ldc, int K)
{
    const int wave = threadIdx.x >> 5;
    const int lane = threadIdx.x & 31;
    const int half = lane >> 4;
    const int l15  = lane & 15;
    const int mbase = blockIdx.y * 16;
    const int nbase = blockIdx.x * 64 + wave * 16;

    const unsigned short* __restrict__ arow = A  + (size_t)(mbase + l15) * lda;
    const unsigned short* __restrict__ brow = Bt + (size_t)(nbase + l15) * ldb;

    v8f acc = {};
    for (int kb = 0; kb < K; kb += 32) {
        Frag a, b;
#pragma unroll
        for (int j = 0; j < 8; ++j) {
            // 16-bit A-matrix 16x32 VGPR layout (ISA 7.12.2):
            // lanes 0-15: VGPR0-3 => K=0..7, VGPR4-7 => K=16..23
            // lanes 16-31: VGPR0-3 => K=8..15, VGPR4-7 => K=24..31
            const int kk = (j < 4) ? ((half ? 8 : 0)  + 2 * j)
                                   : ((half ? 24 : 16) + 2 * (j - 4));
            a.u[j] = *(const unsigned int*)(arow + kb + kk);
            b.u[j] = *(const unsigned int*)(brow + kb + kk);
        }
        acc = __builtin_amdgcn_wmma_f32_16x16x32_bf16(
            false, a.v, false, b.v, (short)0, acc, false, false);
    }

    const int col = nbase + l15;
    const float bs = bias ? bias[col] : 0.0f;
#pragma unroll
    for (int r = 0; r < 8; ++r) {
        const int row = mbase + r + (half ? 8 : 0);   // C layout (ISA 7.12.2)
        float v = acc[r] + bs;
        if (ACT == 1) {
            const float x3 = v * v * v;
            v = 0.5f * v * (1.0f + tanhf(0.7978845608028654f * (v + 0.044715f * x3)));
        }
        if (Cin) v += Cin[(size_t)row * ldc + col];
        if (Cf)  Cf[(size_t)row * ldc + col] = v;
        if (Cb)  Cb[(size_t)row * ldc + col] = f2bf(v);
    }
}

// ---------------------------------------------------------------------------
// Mean-pool over axis 1: x (128,512,1024) f32 -> bf16 (128,1024).
// float4 loads, fully coalesced; this is the only real HBM traffic (~256MB).
// ---------------------------------------------------------------------------
__global__ __launch_bounds__(256)
void pool_kernel(const float* __restrict__ x, unsigned short* __restrict__ out)
{
    const int b = blockIdx.x;                 // 128 blocks
    const int c4 = threadIdx.x;               // 256 threads * 4 cols = 1024
    const float4* __restrict__ p = (const float4*)(x + (size_t)b * 512 * 1024) + c4;
    float4 s = make_float4(0.f, 0.f, 0.f, 0.f);
    for (int t = 0; t < 512; ++t) {
        const float4 v = p[(size_t)t * 256];
        s.x += v.x; s.y += v.y; s.z += v.z; s.w += v.w;
    }
    const float inv = 1.0f / 512.0f;
    unsigned short* o = out + (size_t)b * 1024 + c4 * 4;
    o[0] = f2bf(s.x * inv); o[1] = f2bf(s.y * inv);
    o[2] = f2bf(s.z * inv); o[3] = f2bf(s.w * inv);
}

// ---------------------------------------------------------------------------
// Transpose + f32->bf16: dst[c*R + r] = bf16(src[r*ld + c]); src is R x C.
// ---------------------------------------------------------------------------
__global__ __launch_bounds__(256)
void transpose_conv(const float* __restrict__ src, int ld,
                    unsigned short* __restrict__ dst, int R, int C)
{
    __shared__ float tile[32][33];
    (void)C;
    const int c0 = blockIdx.x * 32, r0 = blockIdx.y * 32;
    const int tx = threadIdx.x & 31, ty = threadIdx.x >> 5;
#pragma unroll
    for (int i = ty; i < 32; i += 8)
        tile[i][tx] = src[(size_t)(r0 + i) * ld + c0 + tx];
    __syncthreads();
#pragma unroll
    for (int i = ty; i < 32; i += 8)
        dst[(size_t)(c0 + i) * R + r0 + tx] = f2bf(tile[tx][i]);
}

// ---------------------------------------------------------------------------
// LayerNorm over D=512, one row per block, fp32 math, bf16 output.
// ---------------------------------------------------------------------------
__global__ __launch_bounds__(256)
void ln_kernel(const float* __restrict__ x, const float* __restrict__ g,
               const float* __restrict__ b, unsigned short* __restrict__ out)
{
    __shared__ float s1[256], s2[256];
    const int row = blockIdx.x, t = threadIdx.x;
    const float* xr = x + (size_t)row * 512;
    const float a0 = xr[t], a1 = xr[t + 256];
    s1[t] = a0 + a1;
    s2[t] = a0 * a0 + a1 * a1;
    __syncthreads();
    for (int off = 128; off > 0; off >>= 1) {
        if (t < off) { s1[t] += s1[t + off]; s2[t] += s2[t + off]; }
        __syncthreads();
    }
    const float mean = s1[0] * (1.0f / 512.0f);
    const float var  = s2[0] * (1.0f / 512.0f) - mean * mean;
    const float rs   = rsqrtf(var + 1e-5f);
    out[(size_t)row * 512 + t]       = f2bf((a0 - mean) * rs * g[t] + b[t]);
    out[(size_t)row * 512 + t + 256] = f2bf((a1 - mean) * rs * g[t + 256] + b[t + 256]);
}

// ---------------------------------------------------------------------------
// Elementwise utilities
// ---------------------------------------------------------------------------
__global__ void cvt_bf16_kernel(const float* __restrict__ in,
                                unsigned short* __restrict__ out, int n)
{
    int i = blockIdx.x * blockDim.x + threadIdx.x;
    if (i < n) out[i] = f2bf(in[i]);
}
__global__ void add_f32_kernel(float* __restrict__ dst,
                               const float* __restrict__ src, int n)
{
    int i = blockIdx.x * blockDim.x + threadIdx.x;
    if (i < n) dst[i] += src[i];
}
__global__ void zero_f32_kernel(float* __restrict__ p, int n)
{
    int i = blockIdx.x * blockDim.x + threadIdx.x;
    if (i < n) p[i] = 0.0f;
}
__global__ void copy_f32_kernel(float* __restrict__ dst,
                                const float* __restrict__ src, int n)
{
    int i = blockIdx.x * blockDim.x + threadIdx.x;
    if (i < n) dst[i] = src[i];
}

// ---------------------------------------------------------------------------
// Host side
// ---------------------------------------------------------------------------
static inline char* carve(char*& p, size_t bytes) {
    char* r = p;
    p += (bytes + 255) & ~(size_t)255;
    return r;
}

static void launch_gemm(hipStream_t s,
                        const unsigned short* A, int lda,
                        const unsigned short* Bt, int ldb,
                        const float* bias, const float* Cin,
                        float* Cf, unsigned short* Cb,
                        int ldc, int M, int N, int K, int act)
{
    dim3 grid(N / 64, M / 16);
    if (act)
        gemm_bf16_wmma<1><<<grid, 128, 0, s>>>(A, lda, Bt, ldb, bias, Cin, Cf, Cb, ldc, K);
    else
        gemm_bf16_wmma<0><<<grid, 128, 0, s>>>(A, lda, Bt, ldb, bias, Cin, Cf, Cb, ldc, K);
}

extern "C" void kernel_launch(void* const* d_in, const int* in_sizes, int n_in,
                              void* d_out, int out_size, void* d_ws, size_t ws_size,
                              hipStream_t stream)
{
    (void)in_sizes; (void)n_in; (void)out_size; (void)ws_size;
    const float* x_encoded = (const float*)d_in[0];
    const float* y_current = (const float*)d_in[1];
    const float* Wpi   = (const float*)d_in[2];
    const float* bpi   = (const float*)d_in[3];
    const float* Wpa   = (const float*)d_in[4];
    const float* bpa   = (const float*)d_in[5];
    const float* Wc    = (const float*)d_in[6];
    const float* bc    = (const float*)d_in[7];
    const float* ln1_g = (const float*)d_in[8];
    const float* ln1_b = (const float*)d_in[9];
    const float* wqkv  = (const float*)d_in[10];
    const float* bqkv  = (const float*)d_in[11];
    const float* wo    = (const float*)d_in[12];
    const float* bo    = (const float*)d_in[13];
    const float* ln2_g = (const float*)d_in[14];
    const float* ln2_b = (const float*)d_in[15];
    const float* w1f   = (const float*)d_in[16];
    const float* b1f   = (const float*)d_in[17];
    const float* w2f   = (const float*)d_in[18];
    const float* b2f   = (const float*)d_in[19];
    float* out = (float*)d_out;

    typedef unsigned short bf;
    char* p = (char*)d_ws;
    bf* wpiT  = (bf*)carve(p, (size_t)RD * HID * 2);     // [512][1024]
    bf* wpaT  = (bf*)carve(p, (size_t)RD * AD * 2);      // [512][256]
    bf* wcT   = (bf*)carve(p, (size_t)RD * 3 * RD * 2);  // [512][1536]
    bf* wvT[NLAYER], *woT[NLAYER], *w1T[NLAYER], *w2T[NLAYER];
    for (int l = 0; l < NLAYER; ++l) {
        wvT[l] = (bf*)carve(p, (size_t)RD * RD * 2);     // [512][512]
        woT[l] = (bf*)carve(p, (size_t)RD * RD * 2);     // [512][512]
        w1T[l] = (bf*)carve(p, (size_t)FF * RD * 2);     // [2048][512]
        w2T[l] = (bf*)carve(p, (size_t)RD * FF * 2);     // [512][2048]
    }
    bf*    xpool_bf = (bf*)carve(p, (size_t)M_ROWS * HID * 2);
    bf*    y_bf     = (bf*)carve(p, (size_t)M_ROWS * AD * 2);
    bf*    xproj_bf = (bf*)carve(p, (size_t)M_ROWS * RD * 2);
    bf*    yproj_bf = (bf*)carve(p, (size_t)M_ROWS * RD * 2);
    float* base_f   = (float*)carve(p, (size_t)M_ROWS * RD * 4);
    float* z_f      = (float*)carve(p, (size_t)M_ROWS * RD * 4);
    bf*    z_bf     = (bf*)carve(p, (size_t)M_ROWS * RD * 2);
    float* t_f      = (float*)carve(p, (size_t)M_ROWS * RD * 4);
    bf*    h_bf     = (bf*)carve(p, (size_t)M_ROWS * RD * 2);
    bf*    v_bf     = (bf*)carve(p, (size_t)M_ROWS * RD * 2);
    bf*    f_bf     = (bf*)carve(p, (size_t)M_ROWS * FF * 2);

    const int NZ = M_ROWS * RD;   // 65536

    // --- Stage 0: pooling (HBM bound) + input conversions ------------------
    pool_kernel<<<M_ROWS, 256, 0, stream>>>(x_encoded, xpool_bf);
    cvt_bf16_kernel<<<(M_ROWS * AD + 255) / 256, 256, 0, stream>>>(y_current, y_bf, M_ROWS * AD);

    // --- Stage 1: weight transpose + bf16 conversion (L2 resident) --------
    transpose_conv<<<dim3(RD / 32, HID / 32), 256, 0, stream>>>(Wpi, RD, wpiT, HID, RD);
    transpose_conv<<<dim3(RD / 32, AD / 32), 256, 0, stream>>>(Wpa, RD, wpaT, AD, RD);
    transpose_conv<<<dim3(RD / 32, (3 * RD) / 32), 256, 0, stream>>>(Wc, RD, wcT, 3 * RD, RD);
    for (int l = 0; l < NLAYER; ++l) {
        // v-slice of wqkv only (softmax over S=1 is identity => q,k are dead)
        transpose_conv<<<dim3(RD / 32, RD / 32), 256, 0, stream>>>(
            wqkv + (size_t)l * RD * 3 * RD + 2 * RD, 3 * RD, wvT[l], RD, RD);
        transpose_conv<<<dim3(RD / 32, RD / 32), 256, 0, stream>>>(
            wo + (size_t)l * RD * RD, RD, woT[l], RD, RD);
        transpose_conv<<<dim3(FF / 32, RD / 32), 256, 0, stream>>>(
            w1f + (size_t)l * RD * FF, FF, w1T[l], RD, FF);
        transpose_conv<<<dim3(RD / 32, FF / 32), 256, 0, stream>>>(
            w2f + (size_t)l * FF * RD, RD, w2T[l], FF, RD);
    }

    // --- Stage 2: loop-invariant projections -------------------------------
    launch_gemm(stream, xpool_bf, HID, wpiT, HID, bpi, nullptr, nullptr, xproj_bf,
                RD, M_ROWS, RD, HID, 0);
    launch_gemm(stream, y_bf, AD, wpaT, AD, bpa, nullptr, nullptr, yproj_bf,
                RD, M_ROWS, RD, AD, 0);
    // base = x_proj @ Wc[0:512] + y_proj @ Wc[512:1024] + bc
    launch_gemm(stream, xproj_bf, RD, wcT + 0, 3 * RD, bc, nullptr, base_f, nullptr,
                RD, M_ROWS, RD, RD, 0);
    launch_gemm(stream, yproj_bf, RD, wcT + RD, 3 * RD, nullptr, base_f, base_f, nullptr,
                RD, M_ROWS, RD, RD, 0);

    zero_f32_kernel<<<(NZ + 255) / 256, 256, 0, stream>>>(z_f, NZ);

    // --- Stage 3: recursive transformer ------------------------------------
    for (int it = 0; it < NREC; ++it) {
        cvt_bf16_kernel<<<(NZ + 255) / 256, 256, 0, stream>>>(z_f, z_bf, NZ);
        // t = base + z @ Wc[1024:1536]
        launch_gemm(stream, z_bf, RD, wcT + 2 * RD, 3 * RD, nullptr, base_f, t_f, nullptr,
                    RD, M_ROWS, RD, RD, 0);
        for (int l = 0; l < NLAYER; ++l) {
            // attention with S=1: o = v = ln1(t) @ Wv + bv
            ln_kernel<<<M_ROWS, 256, 0, stream>>>(t_f, ln1_g + l * RD, ln1_b + l * RD, h_bf);
            launch_gemm(stream, h_bf, RD, wvT[l], RD, bqkv + (size_t)l * 3 * RD + 2 * RD,
                        nullptr, nullptr, v_bf, RD, M_ROWS, RD, RD, 0);
            launch_gemm(stream, v_bf, RD, woT[l], RD, bo + l * RD, t_f, t_f, nullptr,
                        RD, M_ROWS, RD, RD, 0);
            // FFN with fused tanh-GELU
            ln_kernel<<<M_ROWS, 256, 0, stream>>>(t_f, ln2_g + l * RD, ln2_b + l * RD, h_bf);
            launch_gemm(stream, h_bf, RD, w1T[l], RD, b1f + (size_t)l * FF,
                        nullptr, nullptr, f_bf, FF, M_ROWS, FF, RD, 1);
            launch_gemm(stream, f_bf, FF, w2T[l], FF, b2f + l * RD, t_f, t_f, nullptr,
                        RD, M_ROWS, RD, FF, 0);
        }
        add_f32_kernel<<<(NZ + 255) / 256, 256, 0, stream>>>(z_f, t_f, NZ);
    }

    copy_f32_kernel<<<(NZ + 255) / 256, 256, 0, stream>>>(out, z_f, NZ);
}
